// TM_20005957665089
// MI455X (gfx1250) — compile-verified, matrix operand order checked by reading, not run on hardware
//
#include <hip/hip_runtime.h>
#include <cstdint>

typedef __attribute__((ext_vector_type(16))) _Float16 v16h;
typedef __attribute__((ext_vector_type(8)))  float    v8f;

#define NIMG   8192
#define NPIX   81
#define NP     96      // 6 N-tiles of 16
#define KC     64      // K slots in bcat: [0,16)=q, [16,48)=f, [48,64)=0
#define FDIM   2592    // 32*81
#define NCLS   10
#define MAXDEPTH 8
#define CONF_TH 0.9f

union FragH { v16h v; uint32_t u[8]; _Float16 h[16]; };
union Pack8 { uint4 q; uint32_t u[4]; _Float16 h[8]; };

// Load a 16x32 f16 WMMA A/B fragment from a row-major [row][k] buffer
// (k contiguous). `row` must already include the per-lane row (lane&15 + base).
// ISA layout: lanes 0-15 hold K pairs {0..7,16..23}, lanes 16-31 hold {8..15,24..31}.
__device__ __forceinline__ v16h load_frag(const _Float16* base, int row, int stride,
                                          int k0, int lane) {
  const _Float16* rp = base + row * stride;
  int kb = k0 + ((lane & 16) ? 8 : 0);
  FragH f;
#pragma unroll
  for (int j = 0; j < 8; ++j) {
    int k = kb + ((j & 4) ? 16 : 0) + ((j & 3) << 1);
    f.u[j] = *reinterpret_cast<const uint32_t*>(rp + k);
  }
  return f.v;
}

__device__ __forceinline__ v8f wmma_f16(v16h a, v16h b, v8f c) {
  return __builtin_amdgcn_wmma_f32_16x16x32_f16(false, a, false, b, (short)0, c,
                                                false, false);
}

// Async copy 16B/lane global -> LDS (GLOBAL_LOAD_ASYNC_TO_LDS_B128, ASYNCcnt).
__device__ __forceinline__ void async_load_b128(void* lds, const void* g) {
  uint32_t l = (uint32_t)(uintptr_t)lds;
  uint64_t ga = (uint64_t)(uintptr_t)g;
  asm volatile("global_load_async_to_lds_b128 %0, %1, off"
               :: "v"(l), "v"(ga) : "memory");
}
__device__ __forceinline__ void wait_async0() {
  asm volatile("s_wait_asynccnt 0x0" ::: "memory");
}

// ---------------------------------------------------------------------------
// Kernel 1: base module, WMMA im2col formulation. One block (4 waves)/image.
// conv1: M=8(pad16) K=75(pad96) N=900, 15 strips of 60 positions (2 out rows)
// conv2: M=16       K=72(pad96) N=225,  4 strips of 60/45 positions
// Strips align with the 2x2 maxpool so pooling is strip-local.
// Input image staged via global_load_async_to_lds_b128; bias seeds in C-frags.
// ---------------------------------------------------------------------------
__global__ __launch_bounds__(128) void base_kernel(
    const float* __restrict__ x,
    const float* __restrict__ w1, const float* __restrict__ b1,
    const float* __restrict__ w2, const float* __restrict__ b2,
    const float* __restrict__ w3, const float* __restrict__ b3,
    _Float16* __restrict__ F0) {
  __shared__ alignas(16) float    xf32[3 * 32 * 32];   // 12.3 KB async-staged
  __shared__ alignas(16) _Float16 xh[3 * 32 * 32];     // 6 KB
  __shared__ alignas(16) _Float16 w1h[16 * 96];        // 3 KB  (rows 8..15 zero)
  __shared__ alignas(16) _Float16 w2h[16 * 96];        // 3 KB
  __shared__ alignas(16) _Float16 imcol[64 * 96];      // 12.3 KB im2col strip
  __shared__ alignas(16) _Float16 cst[64 * 16];        // 2 KB conv-out staging
  __shared__ alignas(16) _Float16 p1h[8 * 225];        // pooled1 [8][15][15]
  __shared__ alignas(16) _Float16 p2h[16 * 49];        // pooled2 [16][7][7]
  __shared__ float w3s[512];
  __shared__ float b3s[32];

  const int tid = threadIdx.x;
  const int lane = tid & 31;
  const int wv = tid >> 5;                 // 4 waves
  const int img = blockIdx.x;
  const float* xg = x + (size_t)img * 3072;

  // issue async input staging first (overlaps with weight staging below)
  for (int i = tid * 4; i < 3072; i += 512) async_load_b128(&xf32[i], xg + i);

  // bias accumulator seeds (C-fragment layout: c[r] = bias[mb + r])
  const int mbh = (lane & 16) ? 8 : 0;
  v8f C1s, C2s;
#pragma unroll
  for (int r = 0; r < 8; ++r) {
    int m = mbh + r;
    C1s[r] = (m < 8) ? b1[m] : 0.f;
    C2s[r] = b2[m];
  }

  for (int i = tid; i < 16 * 96; i += 128) {
    int m = i / 96, k = i % 96;
    w1h[i] = (m < 8 && k < 75) ? (_Float16)w1[m * 75 + k] : (_Float16)0.f;
  }
  for (int i = tid; i < 16 * 96; i += 128) {
    int m = i / 96, k = i % 96;
    w2h[i] = (k < 72) ? (_Float16)w2[m * 72 + k] : (_Float16)0.f;
  }
  for (int i = tid; i < 512; i += 128) w3s[i] = w3[i];
  if (tid < 32) b3s[tid] = b3[tid];
  wait_async0();
  __syncthreads();
  for (int i = tid; i < 3072; i += 128) xh[i] = (_Float16)xf32[i];
  __syncthreads();

  // loop-invariant weight A fragments
  v16h A1[3], A2[3];
#pragma unroll
  for (int ks = 0; ks < 3; ++ks) {
    A1[ks] = load_frag(w1h, lane & 15, 96, 32 * ks, lane);
    A2[ks] = load_frag(w2h, lane & 15, 96, 32 * ks, lane);
  }

  // ---- conv1 (5x5 pad1) + relu + 2x2 pool, strip = 2 conv rows (60 pos) ----
  for (int py = 0; py < 15; ++py) {
    for (int i = tid; i < 64 * 96; i += 128) {
      int n = i / 96, k = i % 96;
      _Float16 v = (_Float16)0.f;
      if (n < 60 && k < 75) {
        int dy = n / 30, ox = n % 30, oy = 2 * py + dy;
        int c = k / 25, r = k % 25, ky = r / 5, kx = r % 5;
        int iy = oy - 1 + ky, ix = ox - 1 + kx;
        if (iy >= 0 && iy < 32 && ix >= 0 && ix < 32) v = xh[c * 1024 + iy * 32 + ix];
      }
      imcol[i] = v;
    }
    __syncthreads();
    {
      int nb = 16 * wv + (lane & 15);
      v8f c = C1s;
#pragma unroll
      for (int ks = 0; ks < 3; ++ks)
        c = wmma_f16(A1[ks], load_frag(imcol, nb, 96, 32 * ks, lane), c);
      Pack8 p;
#pragma unroll
      for (int r = 0; r < 8; ++r) p.h[r] = (_Float16)fmaxf(c[r], 0.f);
      *reinterpret_cast<uint4*>(&cst[nb * 16 + mbh]) = p.q;  // b128 LDS store
    }
    __syncthreads();
    if (tid < 120) {  // 8 ch x 15 pooled cols
      int o = tid / 15, px = tid % 15;
      float v0 = (float)cst[(2 * px) * 16 + o];
      float v1 = (float)cst[(2 * px + 1) * 16 + o];
      float v2 = (float)cst[(30 + 2 * px) * 16 + o];
      float v3 = (float)cst[(31 + 2 * px) * 16 + o];
      p1h[o * 225 + py * 15 + px] = (_Float16)fmaxf(fmaxf(v0, v1), fmaxf(v2, v3));
    }
    __syncthreads();
  }

  // ---- conv2 (3x3 pad1) + relu + pool, strip = 4 conv rows (2 pool rows) ----
  for (int sy = 0; sy < 4; ++sy) {
    int nvalid = (sy < 3) ? 60 : 45;
    for (int i = tid; i < 64 * 96; i += 128) {
      int n = i / 96, k = i % 96;
      _Float16 v = (_Float16)0.f;
      if (n < nvalid && k < 72) {
        int oy = 4 * sy + n / 15, ox = n % 15;
        int c = k / 9, r = k % 9, ky = r / 3, kx = r % 3;
        int iy = oy - 1 + ky, ix = ox - 1 + kx;
        if (iy >= 0 && iy < 15 && ix >= 0 && ix < 15) v = p1h[c * 225 + iy * 15 + ix];
      }
      imcol[i] = v;
    }
    __syncthreads();
    {
      int nb = 16 * wv + (lane & 15);
      v8f c = C2s;
#pragma unroll
      for (int ks = 0; ks < 3; ++ks)
        c = wmma_f16(A2[ks], load_frag(imcol, nb, 96, 32 * ks, lane), c);
      Pack8 p;
#pragma unroll
      for (int r = 0; r < 8; ++r) p.h[r] = (_Float16)fmaxf(c[r], 0.f);
      *reinterpret_cast<uint4*>(&cst[nb * 16 + mbh]) = p.q;
    }
    __syncthreads();
    if (tid < 224) {  // 16 ch x 2 pool rows x 7 pool cols
      int o = tid / 14, r2 = tid % 14;
      int pyl = r2 / 7, px = r2 % 7;
      int py = 2 * sy + pyl;
      if (py < 7) {
        int n00 = (2 * pyl) * 15 + 2 * px;
        float v0 = (float)cst[n00 * 16 + o];
        float v1 = (float)cst[(n00 + 1) * 16 + o];
        float v2 = (float)cst[(n00 + 15) * 16 + o];
        float v3 = (float)cst[(n00 + 16) * 16 + o];
        p2h[o * 49 + py * 7 + px] = (_Float16)fmaxf(fmaxf(v0, v1), fmaxf(v2, v3));
      }
    }
    __syncthreads();
  }

  // ---- conv3 (1x1, 16->32) into zero-padded 9x9, stored f16 ----
  for (int idx = tid; idx < FDIM; idx += 128) {
    int c = idx / NPIX, p = idx % NPIX;
    int y = p / 9, xx = p % 9;
    float v = 0.f;
    if (y < 7 && xx < 7) {
      v = b3s[c];
      for (int i = 0; i < 16; ++i) v += (float)p2h[i * 49 + y * 7 + xx] * w3s[c * 16 + i];
    }
    F0[(size_t)img * FDIM + idx] = (_Float16)v;
  }
}

// ---------------------------------------------------------------------------
// Kernel 2: 8-iteration adaptive-depth loop, all GEMMs via v_wmma.
// 2 waves / block, one image per wave; f resident in LDS for all iterations.
// bcat layout: [pixel n (96)][K slot (64)]  k=0..15 q, 16..47 f, 48..63 zero.
// F0 tile async-bulk-copied to LDS (scratch), transposed locally; scratch is
// then reused as the conf-activation buffer (disjoint lifetimes).
// ---------------------------------------------------------------------------
__global__ __launch_bounds__(64) void loop_kernel(
    const _Float16* __restrict__ F0,
    const float* __restrict__ wcf_g, const float* __restrict__ bcf_g,
    const float* __restrict__ wcfc_g, const float* __restrict__ bcfc_g,
    const float* __restrict__ wq_g, const float* __restrict__ bq_g,
    const float* __restrict__ wa_g, const float* __restrict__ ba_g,
    _Float16* __restrict__ Ff) {
  __shared__ alignas(16) _Float16 bcat[2 * NP * KC];     // 24.0 KB
  __shared__ alignas(16) _Float16 scratch[2 * NP * KC];  // 24.0 KB (F0 stage / conf h)
  __shared__ alignas(16) _Float16 wcf_h[64 * 32];        // 4 KB
  __shared__ alignas(16) _Float16 wq_h[16 * 32];         // 1 KB
  __shared__ alignas(16) _Float16 wa_h[32 * 64];         // 4 KB (K padded 48->64)
  __shared__ float wcfc[1024];                            // 4 KB

  const int tid = threadIdx.x;
  const int wave = tid >> 5;
  const int lane = tid & 31;
  const int img = blockIdx.x * 2 + wave;

  _Float16* bc = bcat + wave * NP * KC;
  _Float16* scr = scratch + wave * NP * KC;

  // async bulk copy of this image's F0 tile (2592 f16 = 324 x 16B) into LDS
  {
    const _Float16* fg = F0 + (size_t)img * FDIM;
    for (int ch = lane; ch < 324; ch += 32)
      async_load_b128(&scr[ch * 8], fg + ch * 8);
  }

  // stage weights (block-cooperative) while async copies fly
  for (int i = tid; i < 2048; i += 64) wcf_h[i] = (_Float16)wcf_g[i];
  for (int i = tid; i < 512; i += 64) wq_h[i] = (_Float16)wq_g[i];
  for (int i = tid; i < 2048; i += 64) {
    int m = i >> 6, k = i & 63;
    wa_h[i] = (k < 48) ? (_Float16)wa_g[m * 48 + k] : (_Float16)0.f;
  }
  for (int i = tid; i < 1024; i += 64) wcfc[i] = wcfc_g[i];

  // bias accumulator seeds in registers (C-fragment layout)
  const int mbh = (lane & 16) ? 8 : 0;
  v8f Ccf[4], Cq, Ca[2];
#pragma unroll
  for (int r = 0; r < 8; ++r) {
#pragma unroll
    for (int mt = 0; mt < 4; ++mt) Ccf[mt][r] = bcf_g[16 * mt + mbh + r];
    Cq[r] = bq_g[mbh + r];
#pragma unroll
    for (int mt = 0; mt < 2; ++mt) Ca[mt][r] = ba_g[16 * mt + mbh + r];
  }

  wait_async0();
  // LDS-local transpose [c][p] -> bcat[n][k], q rows + pads zeroed (b128 stores)
  for (int n = lane; n < NP; n += 32) {
#pragma unroll
    for (int kc0 = 0; kc0 < KC; kc0 += 8) {
      Pack8 p;
#pragma unroll
      for (int j = 0; j < 8; ++j) {
        int k = kc0 + j;
        _Float16 v = (_Float16)0.f;
        if (k >= 16 && k < 48 && n < NPIX) v = scr[(k - 16) * NPIX + n];
        p.h[j] = v;
      }
      *reinterpret_cast<uint4*>(&bc[n * KC + kc0]) = p.q;
    }
  }
  __syncthreads();

  _Float16* hw = scr;  // scratch reused as conf-activation buffer from here on

  v16h Acf[4], Aq, Aa[2][2];
#pragma unroll
  for (int mt = 0; mt < 4; ++mt)
    Acf[mt] = load_frag(wcf_h, 16 * mt + (lane & 15), 32, 0, lane);
  Aq = load_frag(wq_h, (lane & 15), 32, 0, lane);
#pragma unroll
  for (int mt = 0; mt < 2; ++mt)
#pragma unroll
    for (int ks = 0; ks < 2; ++ks)
      Aa[mt][ks] = load_frag(wa_h, 16 * mt + (lane & 15), 64, 32 * ks, lane);

  bool active = true;
  const float bcfc0 = bcfc_g[0];

  for (int it = 0; it < MAXDEPTH; ++it) {
    // ---- conf conv 1x1: h = relu(Wcf[64,32] @ f[32,81]) -> hw ----
#pragma unroll
    for (int nt = 0; nt < 6; ++nt) {
      int n = 16 * nt + (lane & 15);
      v16h bf = load_frag(bc, n, KC, 16, lane);
#pragma unroll
      for (int mt = 0; mt < 4; ++mt) {
        v8f c = Ccf[mt];
        c = wmma_f16(Acf[mt], bf, c);
        Pack8 p;
#pragma unroll
        for (int r = 0; r < 8; ++r) p.h[r] = (_Float16)fmaxf(c[r], 0.f);
        *reinterpret_cast<uint4*>(&hw[n * KC + 16 * mt + mbh]) = p.q;
      }
    }
    __syncthreads();

    // ---- 2x2 maxpool (9x9->4x4) + 1024->1 FC + sigmoid, wave reduce ----
    {
      int pc = lane & 15;
      int py = pc >> 2, px = pc & 3;
      int p00 = (2 * py) * 9 + 2 * px;
      int ch0 = (lane >> 4) * 32;
      float acc = 0.f;
#pragma unroll
      for (int cc = 0; cc < 32; ++cc) {
        int c2 = ch0 + cc;
        float a0 = (float)hw[p00 * KC + c2];
        float a1 = (float)hw[(p00 + 1) * KC + c2];
        float a2 = (float)hw[(p00 + 9) * KC + c2];
        float a3 = (float)hw[(p00 + 10) * KC + c2];
        acc += fmaxf(fmaxf(a0, a1), fmaxf(a2, a3)) * wcfc[c2 * 16 + pc];
      }
#pragma unroll
      for (int off = 16; off > 0; off >>= 1) acc += __shfl_xor(acc, off, 32);
      float confv = 1.f / (1.f + expf(-(acc + bcfc0)));
      active = active && (confv < CONF_TH);
    }

    // ---- q conv 1x1: q = Wq[16,32] @ f -> bcat rows 0..15 ----
#pragma unroll
    for (int nt = 0; nt < 6; ++nt) {
      int n = 16 * nt + (lane & 15);
      v16h bf = load_frag(bc, n, KC, 16, lane);
      v8f c = Cq;
      c = wmma_f16(Aq, bf, c);
      Pack8 p;
#pragma unroll
      for (int r = 0; r < 8; ++r) p.h[r] = (_Float16)c[r];
      *reinterpret_cast<uint4*>(&bc[n * KC + mbh]) = p.q;
    }
    __syncthreads();

    // ---- a = tanh(Wa[32,48] @ [q;f][48,81]); f += a where active ----
#pragma unroll
    for (int nt = 0; nt < 6; ++nt) {
      int n = 16 * nt + (lane & 15);
      bool nok = (n < NPIX);
      v16h b0 = load_frag(bc, n, KC, 0, lane);
      v16h b1v = load_frag(bc, n, KC, 32, lane);
#pragma unroll
      for (int mt = 0; mt < 2; ++mt) {
        v8f c = Ca[mt];
        c = wmma_f16(Aa[mt][0], b0, c);
        c = wmma_f16(Aa[mt][1], b1v, c);
        // b128 read-modify-write of the f rows, predicated on active & n<81
        Pack8 p;
        p.q = *reinterpret_cast<const uint4*>(&bc[n * KC + 16 + 16 * mt + mbh]);
#pragma unroll
        for (int r = 0; r < 8; ++r)
          p.h[r] = (_Float16)((float)p.h[r] + tanhf(c[r]));
        if (active && nok)
          *reinterpret_cast<uint4*>(&bc[n * KC + 16 + 16 * mt + mbh]) = p.q;
      }
    }
    __syncthreads();
  }

  for (int idx = lane; idx < FDIM; idx += 32) {
    int c = idx / NPIX, p = idx % NPIX;
    Ff[(size_t)img * FDIM + idx] = bc[p * KC + 16 + c];
  }
}

// ---------------------------------------------------------------------------
// Kernel 3: final classifier GEMM out[8192,10] = F[8192,2592] @ Wcls^T + b.
// 4 waves/block, 16 images per wave, K = 81 steps of 32, Wcls tile via LDS.
// ---------------------------------------------------------------------------
__global__ __launch_bounds__(128) void fc_kernel(
    const _Float16* __restrict__ Ff,
    const float* __restrict__ wcls, const float* __restrict__ bcls,
    float* __restrict__ out) {
  __shared__ alignas(16) _Float16 Bls[16 * 32];  // [n][k], classes padded to 16

  const int tid = threadIdx.x;
  const int wave = tid >> 5;
  const int lane = tid & 31;
  const int tile = blockIdx.x * 4 + wave;
  const int imgBase = tile * 16;
  const _Float16* arow = Ff + (size_t)(imgBase + (lane & 15)) * FDIM;

  int n = lane & 15;
  float bias = (n < NCLS) ? bcls[n] : 0.f;
  v8f c;
#pragma unroll
  for (int r = 0; r < 8; ++r) c[r] = bias;  // seed accumulator with bias

  for (int s = 0; s < 81; ++s) {
    int k0 = s * 32;
    __syncthreads();
#pragma unroll
    for (int j = 0; j < 4; ++j) {
      int idx = tid * 4 + j;
      int nn = idx >> 5, kk = idx & 31;
      Bls[nn * 32 + kk] = (nn < NCLS) ? (_Float16)wcls[(size_t)nn * FDIM + k0 + kk]
                                      : (_Float16)0.f;
    }
    __syncthreads();
    if (s + 2 < 81) __builtin_prefetch(arow + k0 + 64, 0, 1);  // global_prefetch_b8
    v16h af = load_frag(arow, 0, 0, k0, lane);
    v16h bf = load_frag(Bls, (lane & 15), 32, 0, lane);
    c = wmma_f16(af, bf, c);
  }

  int mb = (lane & 16) ? 8 : 0;
  if (n < NCLS) {
#pragma unroll
    for (int r = 0; r < 8; ++r)
      out[(size_t)(imgBase + mb + r) * NCLS + n] = c[r];
  }
}

// ---------------------------------------------------------------------------
extern "C" void kernel_launch(void* const* d_in, const int* in_sizes, int n_in,
                              void* d_out, int out_size, void* d_ws, size_t ws_size,
                              hipStream_t stream) {
  const float* x     = (const float*)d_in[0];
  const float* w_b1  = (const float*)d_in[1];
  const float* b_b1  = (const float*)d_in[2];
  const float* w_b2  = (const float*)d_in[3];
  const float* b_b2  = (const float*)d_in[4];
  const float* w_b3  = (const float*)d_in[5];
  const float* b_b3  = (const float*)d_in[6];
  const float* w_cf  = (const float*)d_in[7];
  const float* b_cf  = (const float*)d_in[8];
  const float* w_cfc = (const float*)d_in[9];
  const float* b_cfc = (const float*)d_in[10];
  const float* w_q   = (const float*)d_in[11];
  const float* b_q   = (const float*)d_in[12];
  const float* w_a   = (const float*)d_in[13];
  const float* b_a   = (const float*)d_in[14];
  const float* w_cls = (const float*)d_in[15];
  const float* b_cls = (const float*)d_in[16];
  float* out = (float*)d_out;

  size_t fbytes = (size_t)NIMG * FDIM * sizeof(_Float16);
  _Float16* F0 = (_Float16*)d_ws;
  _Float16* Ff = (_Float16*)((char*)d_ws + ((fbytes + 255) & ~(size_t)255));

  base_kernel<<<NIMG, 128, 0, stream>>>(x, w_b1, b_b1, w_b2, b_b2, w_b3, b_b3, F0);
  loop_kernel<<<NIMG / 2, 64, 0, stream>>>(F0, w_cf, b_cf, w_cfc, b_cfc,
                                           w_q, b_q, w_a, b_a, Ff);
  fc_kernel<<<NIMG / 64, 128, 0, stream>>>(Ff, w_cls, b_cls, out);
}
